// CombineGraph_88330297409791
// MI455X (gfx1250) — compile-verified, matrix operand order checked by reading, not run on hardware
//
#include <hip/hip_runtime.h>

typedef __attribute__((ext_vector_type(16))) __bf16 v16bf;
typedef __attribute__((ext_vector_type(8)))  __bf16 v8bf;
typedef __attribute__((ext_vector_type(8)))  float  v8f;

#define HID 256
#define NT 16            // 256/16 N-tiles
#define ROWS_PER_BLK 128
#define LDS_ROW 40       // 32 bf16 + 8 pad (80B row stride, 16B aligned)

// ---------------- helpers ----------------
__device__ __forceinline__ unsigned lds_off(const void* p) {
    // generic LDS pointer: low 32 bits are the LDS byte offset (ISA 10.2 aperture rules)
    return (unsigned)(uintptr_t)p;
}

__device__ __forceinline__ void async_cp16(unsigned lds, const void* gptr) {
    asm volatile("global_load_async_to_lds_b128 %0, %1, off"
                 :: "v"(lds), "v"((unsigned long long)(uintptr_t)gptr)
                 : "memory");
}

__device__ __forceinline__ void atomic_add_f32(float* p, float v) {
    asm volatile("global_atomic_add_f32 %0, %1, off"
                 :: "v"((unsigned long long)(uintptr_t)p), "v"(v)
                 : "memory");
}

// load one B fragment (32B per lane) from LDS frag-ordered stage
__device__ __forceinline__ v16bf ldb(const __bf16* Bs, int nt, int lane) {
    const v8bf* p = (const v8bf*)(Bs + ((nt * 32 + lane) << 4));
    v8bf lo = p[0], hi = p[1];
    v16bf b;
#pragma unroll
    for (int i = 0; i < 8; ++i) { b[i] = lo[i]; b[i + 8] = hi[i]; }
    return b;
}

__device__ __forceinline__ v16bf lda(const __bf16* As, int arow, int half) {
    v8bf lo = *(const v8bf*)(As + arow * LDS_ROW + half * 8);
    v8bf hi = *(const v8bf*)(As + arow * LDS_ROW + 16 + half * 8);
    v16bf a;
#pragma unroll
    for (int i = 0; i < 8; ++i) { a[i] = lo[i]; a[i + 8] = hi[i]; }
    return a;
}

// issue async copy of one 16KB frag-ordered B stage into LDS (4 x b128 per thread)
__device__ __forceinline__ void issueB(const __bf16* wfrag, unsigned ldsbase,
                                       int kt, int tid) {
    unsigned lds = ldsbase + tid * 64;
    const char* gp = (const char*)wfrag + (size_t)kt * 16384 + tid * 64;
#pragma unroll
    for (int i = 0; i < 4; ++i) async_cp16(lds + i * 16, gp + i * 16);
}

// ---------------- weight pre-swizzle into WMMA B-fragment order ----------------
// frag element (kt, nt, lane, j) = W[kt*32 + kmap(j, lane>>4)][nt*16 + (lane&15)]
__device__ __forceinline__ int kmap(int j, int half) {
    return (j < 8) ? (half * 8 + j) : (16 + half * 8 + (j - 8));
}

__global__ void k_prep_wcat(const float* __restrict__ wpa,
                            const float* __restrict__ wpb,
                            __bf16* __restrict__ wfrag) {
    int idx = blockIdx.x * 256 + threadIdx.x;       // 20*16*32*16 = 163840
    if (idx >= 20 * 16 * 32 * 16) return;
    int j    = idx & 15;
    int lane = (idx >> 4) & 31;
    int nt   = (idx >> 9) & 15;
    int kt   = idx >> 13;
    int k    = kt * 32 + kmap(j, lane >> 4);
    int col  = nt * 16 + (lane & 15);
    float v  = (k < 256) ? wpa[k * HID + col] : wpb[(k - 256) * HID + col];
    wfrag[idx] = (__bf16)v;
}

__global__ void k_prep_wr(const float* __restrict__ wr,
                          __bf16* __restrict__ wfrag) {
    int idx = blockIdx.x * 256 + threadIdx.x;       // 3*8*16*32*16 = 196608
    if (idx >= 3 * 8 * 16 * 32 * 16) return;
    int j    = idx & 15;
    int lane = (idx >> 4) & 31;
    int nt   = (idx >> 9) & 15;
    int kt   = (idx >> 13) & 7;
    int rel  = idx >> 16;
    int k    = kt * 32 + kmap(j, lane >> 4);
    int col  = nt * 16 + (lane & 15);
    wfrag[idx] = (__bf16)wr[((size_t)rel * HID + k) * HID + col];
}

// ---------------- h0 = 0.5*(feat_a@Wp_a + feat_b@Wp_b), row0 zeroed ----------------
__global__ __launch_bounds__(256) void k_h0(const float* __restrict__ fa,
                                            const float* __restrict__ fb,
                                            const __bf16* __restrict__ wfrag,
                                            float* __restrict__ h0f,
                                            __bf16* __restrict__ h0b, int M) {
    __shared__ __attribute__((aligned(16))) __bf16 As[ROWS_PER_BLK * LDS_ROW];
    __shared__ __attribute__((aligned(16))) __bf16 Bs[2][8192];   // 2 x 16KB
    const int tid  = threadIdx.x;
    const int wave = tid >> 5, lane = tid & 31;
    const int half = lane >> 4, l15 = lane & 15;
    const int rowbase = blockIdx.x * ROWS_PER_BLK;

    v8f acc[NT];
#pragma unroll
    for (int t = 0; t < NT; ++t)
#pragma unroll
        for (int i = 0; i < 8; ++i) acc[t][i] = 0.0f;

    issueB(wfrag, lds_off(&Bs[0][0]), 0, tid);

    for (int kt = 0; kt < 20; ++kt) {
        const float* srcp; int stride, kofs;
        if (kt < 8) { srcp = fa; stride = 256; kofs = kt * 32; }
        else        { srcp = fb; stride = 384; kofs = (kt - 8) * 32; }
        // A stage: 128 rows x 32 f32 -> bf16 LDS (convert on VALU)
        {
            int r  = tid >> 1;
            int c0 = (tid & 1) * 16;
            int grow = rowbase + r;
            float vals[16];
            if (grow < M) {
                const float4* gp =
                    (const float4*)(srcp + (size_t)grow * stride + kofs + c0);
#pragma unroll
                for (int i = 0; i < 4; ++i) {
                    float4 t4 = gp[i];
                    vals[4 * i + 0] = t4.x; vals[4 * i + 1] = t4.y;
                    vals[4 * i + 2] = t4.z; vals[4 * i + 3] = t4.w;
                }
            } else {
#pragma unroll
                for (int i = 0; i < 16; ++i) vals[i] = 0.0f;
            }
#pragma unroll
            for (int i = 0; i < 16; ++i)
                As[r * LDS_ROW + c0 + i] = (__bf16)vals[i];
        }
        // prefetch next B stage (wrapped on last iter; drained by s_endpgm)
        issueB(wfrag, lds_off(&Bs[(kt + 1) & 1][0]), (kt + 1) % 20, tid);
        asm volatile("s_wait_asynccnt 0x4" ::: "memory");   // stage kt arrived
        __syncthreads();

        v16bf afrag = lda(As, wave * 16 + l15, half);
        const __bf16* Bsc = &Bs[kt & 1][0];
        v16bf bcur = ldb(Bsc, 0, lane);
#pragma unroll
        for (int nt = 0; nt < NT; ++nt) {
            v16bf bnext = bcur;
            if (nt + 1 < NT) bnext = ldb(Bsc, nt + 1, lane);
            acc[nt] = __builtin_amdgcn_wmma_f32_16x16x32_bf16(
                false, afrag, false, bcur, (short)0, acc[nt], false, false);
            bcur = bnext;
        }
        __syncthreads();
    }

#pragma unroll
    for (int nt = 0; nt < NT; ++nt)
#pragma unroll
        for (int v = 0; v < 8; ++v) {
            int grow = rowbase + wave * 16 + half * 8 + v;
            int col  = nt * 16 + l15;
            float val = 0.5f * acc[nt][v];
            if (grow == 0) val = 0.0f;
            if (grow < M) {
                h0f[(size_t)grow * HID + col] = val;
                h0b[(size_t)grow * HID + col] = (__bf16)val;
            }
        }
}

// ---------------- g = h0_bf16 @ Wr[r]  (bf16 out), fully async-staged ----------------
__device__ __forceinline__ void issueA_gr(const __bf16* A, unsigned ldsbase,
                                          int kofs, int rowbase, int tid, int M) {
    int r = tid >> 1, c0b = (tid & 1) * 32;
    int grow = rowbase + r;
    if (grow < M) {
        unsigned lds = ldsbase + r * (LDS_ROW * 2) + c0b;
        const char* gp = (const char*)A + (size_t)grow * (HID * 2) + kofs * 2 + c0b;
#pragma unroll
        for (int i = 0; i < 2; ++i) async_cp16(lds + i * 16, gp + i * 16);
    }
}

__global__ __launch_bounds__(256) void k_gr(const __bf16* __restrict__ A,
                                            const __bf16* __restrict__ wfrag,
                                            __bf16* __restrict__ g, int M) {
    __shared__ __attribute__((aligned(16))) __bf16 As[2][ROWS_PER_BLK * LDS_ROW];
    __shared__ __attribute__((aligned(16))) __bf16 Bs[2][8192];
    const int tid  = threadIdx.x;
    const int wave = tid >> 5, lane = tid & 31;
    const int half = lane >> 4, l15 = lane & 15;
    const int rowbase = blockIdx.x * ROWS_PER_BLK;

    v8f acc[NT];
#pragma unroll
    for (int t = 0; t < NT; ++t)
#pragma unroll
        for (int i = 0; i < 8; ++i) acc[t][i] = 0.0f;

    issueA_gr(A, lds_off(&As[0][0]), 0, rowbase, tid, M);
    issueB(wfrag, lds_off(&Bs[0][0]), 0, tid);

    for (int kt = 0; kt < 8; ++kt) {
        int nkt = (kt + 1) & 7;
        issueA_gr(A, lds_off(&As[(kt + 1) & 1][0]), nkt * 32, rowbase, tid, M);
        issueB(wfrag, lds_off(&Bs[(kt + 1) & 1][0]), nkt, tid);
        asm volatile("s_wait_asynccnt 0x6" ::: "memory");   // stage kt arrived
        __syncthreads();

        v16bf afrag = lda(&As[kt & 1][0], wave * 16 + l15, half);
        const __bf16* Bsc = &Bs[kt & 1][0];
        v16bf bcur = ldb(Bsc, 0, lane);
#pragma unroll
        for (int nt = 0; nt < NT; ++nt) {
            v16bf bnext = bcur;
            if (nt + 1 < NT) bnext = ldb(Bsc, nt + 1, lane);
            acc[nt] = __builtin_amdgcn_wmma_f32_16x16x32_bf16(
                false, afrag, false, bcur, (short)0, acc[nt], false, false);
            bcur = bnext;
        }
        __syncthreads();
    }

#pragma unroll
    for (int nt = 0; nt < NT; ++nt)
#pragma unroll
        for (int v = 0; v < 8; ++v) {
            int grow = rowbase + wave * 16 + half * 8 + v;
            int col  = nt * 16 + l15;
            if (grow < M) g[(size_t)grow * HID + col] = (__bf16)acc[nt][v];
        }
}

// ---------------- per-relation degree: den[r][dst] += w ----------------
__global__ void k_den(const int* __restrict__ edst, const float* __restrict__ ew,
                      float* __restrict__ den, int N, int E, int EPR) {
    int e = blockIdx.x * 256 + threadIdx.x;
    if (e >= E) return;
    int r = e / EPR;
    atomic_add_f32(den + (size_t)r * N + edst[e], ew[e]);
}

// ---------------- edge scatter: agg[dst] += (w/den[dst]) * g[src] ----------------
__global__ __launch_bounds__(256) void k_edges(const __bf16* __restrict__ g,
                                               const int* __restrict__ esrc,
                                               const int* __restrict__ edst,
                                               const float* __restrict__ ew,
                                               const float* __restrict__ den,
                                               float* __restrict__ agg, int E) {
    int gid  = blockIdx.x * 256 + threadIdx.x;
    int e    = gid >> 5;
    int lane = gid & 31;
    if (e >= E) return;
    int   s  = esrc[e];
    int   d  = edst[e];
    float w  = ew[e];
    float sc = w / fmaxf(den[d], 1e-8f);
    v8bf gv  = *(const v8bf*)(g + (size_t)s * HID + lane * 8);
    float* out = agg + (size_t)d * HID + lane * 8;
#pragma unroll
    for (int i = 0; i < 8; ++i) atomic_add_f32(out + i, sc * (float)gv[i]);
}

// ---------------- residual + LayerNorm, row0 zeroed ----------------
__global__ __launch_bounds__(256) void k_finalize(const float* __restrict__ h0,
                                                  const float* __restrict__ gamma,
                                                  const float* __restrict__ beta,
                                                  float* __restrict__ out, int M) {
    int gid  = blockIdx.x * 256 + threadIdx.x;
    int row  = gid >> 5;
    int lane = gid & 31;
    if (row >= M) return;
    const float4* hp = (const float4*)(h0 + (size_t)row * HID + lane * 8);
    const float4* ap = (const float4*)(out + (size_t)row * HID + lane * 8);
    float4 h0a = hp[0], h0b_ = hp[1], a0 = ap[0], a1 = ap[1];
    float x[8] = { h0a.x + a0.x, h0a.y + a0.y, h0a.z + a0.z, h0a.w + a0.w,
                   h0b_.x + a1.x, h0b_.y + a1.y, h0b_.z + a1.z, h0b_.w + a1.w };
    float s = 0.0f;
#pragma unroll
    for (int i = 0; i < 8; ++i) s += x[i];
#pragma unroll
    for (int off = 16; off > 0; off >>= 1) s += __shfl_xor(s, off, 32);
    float mu = s * (1.0f / HID);
    float vs = 0.0f;
#pragma unroll
    for (int i = 0; i < 8; ++i) { float dx = x[i] - mu; vs += dx * dx; }
#pragma unroll
    for (int off = 16; off > 0; off >>= 1) vs += __shfl_xor(vs, off, 32);
    float rstd = rsqrtf(vs * (1.0f / HID) + 1e-5f);

    float y[8];
#pragma unroll
    for (int i = 0; i < 8; ++i) {
        int col = lane * 8 + i;
        y[i] = (row == 0) ? 0.0f
                          : (x[i] - mu) * rstd * gamma[col] + beta[col];
    }
    float4* op = (float4*)(out + (size_t)row * HID + lane * 8);
    op[0] = make_float4(y[0], y[1], y[2], y[3]);
    op[1] = make_float4(y[4], y[5], y[6], y[7]);
}

extern "C" void kernel_launch(void* const* d_in, const int* in_sizes, int n_in,
                              void* d_out, int out_size, void* d_ws, size_t ws_size,
                              hipStream_t stream) {
    const float* feat_a = (const float*)d_in[0];
    const float* feat_b = (const float*)d_in[1];
    const float* Wp_a   = (const float*)d_in[2];
    const float* Wp_b   = (const float*)d_in[3];
    const float* Wr     = (const float*)d_in[4];
    const float* ln_g   = (const float*)d_in[5];
    const float* ln_b   = (const float*)d_in[6];
    const float* ew     = (const float*)d_in[7];
    const int*   esrc   = (const int*)d_in[8];
    const int*   edst   = (const int*)d_in[9];

    const int M   = in_sizes[0] / HID;   // 100000
    const int E   = in_sizes[7];         // 1500000
    const int EPR = E / 3;

    float* out = (float*)d_out;
    char*  ws  = (char*)d_ws;
    size_t o = 0;
    auto alloc = [&](size_t bytes) {
        size_t p = o;
        o += (bytes + 255) & ~(size_t)255;
        return p;
    };
    __bf16* wcat = (__bf16*)(ws + alloc((size_t)20 * 16 * 32 * 16 * 2));
    __bf16* wrf  = (__bf16*)(ws + alloc((size_t)3 * 8 * 16 * 32 * 16 * 2));
    float*  den  = (float*)(ws + alloc((size_t)3 * M * 4));
    float*  h0f  = (float*)(ws + alloc((size_t)M * HID * 4));
    __bf16* h0b  = (__bf16*)(ws + alloc((size_t)M * HID * 2));
    __bf16* g    = (__bf16*)(ws + alloc((size_t)M * HID * 2));

    hipMemsetAsync(d_out, 0, (size_t)out_size * 4, stream);   // agg accumulator
    hipMemsetAsync(den, 0, (size_t)3 * M * 4, stream);

    k_prep_wcat<<<(20 * 16 * 32 * 16 + 255) / 256, 256, 0, stream>>>(Wp_a, Wp_b, wcat);
    k_prep_wr<<<(3 * 8 * 16 * 32 * 16 + 255) / 256, 256, 0, stream>>>(Wr, wrf);

    int mb = (M + ROWS_PER_BLK - 1) / ROWS_PER_BLK;
    k_h0<<<mb, 256, 0, stream>>>(feat_a, feat_b, wcat, h0f, h0b, M);
    k_den<<<(E + 255) / 256, 256, 0, stream>>>(edst, ew, den, M, E, EPR);

    for (int r = 0; r < 3; ++r) {
        k_gr<<<mb, 256, 0, stream>>>(h0b, wrf + (size_t)r * 8 * 16 * 32 * 16, g, M);
        int eb = (EPR * 32 + 255) / 256;
        k_edges<<<eb, 256, 0, stream>>>(g, esrc + (size_t)r * EPR, edst + (size_t)r * EPR,
                                        ew + (size_t)r * EPR, den + (size_t)r * M, out, EPR);
    }
    k_finalize<<<(M * 32 + 255) / 256, 256, 0, stream>>>(h0f, ln_g, ln_b, out, M);
}
